// Decoder_28321014350363
// MI455X (gfx1250) — matrix operand order repaired and checked by reading, not compiled, not run on hardware
//
#include <hip/hip_runtime.h>
#include <hip/hip_bf16.h>
#include <math.h>

#define NG    10000   // genes
#define NC    8192    // cells
#define ZD    100     // z dim (output cols per gene)
#define GD    100     // g_rep dim
#define KD    32      // key dim  (== WMMA K for f16)
#define HD    256     // hidden dim of key_Z MLP
#define NGPAD 10112   // 79 * 128
#define ZPAD  112     // 7 WMMA N-tiles; col 100 = softmax-denominator column
#define SEG   8       // cell segments (1024 cells each) for parallelism
#define CSEG  (NC / SEG)

typedef _Float16 v16h __attribute__((ext_vector_type(16)));
typedef _Float16 v8h  __attribute__((ext_vector_type(8)));
typedef float    v8f  __attribute__((ext_vector_type(8)));

__device__ __forceinline__ v8f wmma_f16(v16h a, v16h b, v8f c) {
    // D = A(16x32 f16) * B(32x16 f16) + C(16x16 f32)
    return __builtin_amdgcn_wmma_f32_16x16x32_f16(
        false, a, false, b, (short)0, c, false, false);
}

__device__ __forceinline__ float gelu_exact(float x) {
    return 0.5f * x * (1.0f + erff(x * 0.70710678118654752f));
}

// ---------------------------------------------------------------------------
// key = (gelu(raw_Z.T @ Wz1 + bz1) @ Wz2 + bz2)  ->  f16 (NC, KD)
// ---------------------------------------------------------------------------
__global__ __launch_bounds__(HD) void keyz_kernel(
    const float* __restrict__ rawZ,  // (100, NC)
    const float* __restrict__ Wz1,   // (100, HD)
    const float* __restrict__ bz1,
    const float* __restrict__ Wz2,   // (HD, KD)
    const float* __restrict__ bz2,
    _Float16* __restrict__ keyh)     // (NC, KD)
{
    __shared__ float h[HD];
    const int c = blockIdx.x;
    const int j = threadIdx.x;
    float a = bz1[j];
    #pragma unroll 4
    for (int z = 0; z < 100; ++z)
        a = fmaf(rawZ[z * NC + c], Wz1[z * HD + j], a);
    h[j] = gelu_exact(a);
    __syncthreads();
    if (j < KD) {
        float a2 = bz2[j];
        #pragma unroll 8
        for (int jj = 0; jj < HD; ++jj)
            a2 = fmaf(h[jj], Wz2[jj * KD + j], a2);
        keyh[c * KD + j] = (_Float16)a2;
    }
}

// ---------------------------------------------------------------------------
// query = (gelu(G_rep @ Wg1 + bg1) @ Wg2 + bg2) * (1/sqrt(KD))  -> f16
// ---------------------------------------------------------------------------
__global__ __launch_bounds__(KD) void queryg_kernel(
    const float* __restrict__ Grep,  // (NG, 100)
    const float* __restrict__ Wg1,   // (100, KD)
    const float* __restrict__ bg1,
    const float* __restrict__ Wg2,   // (KD, KD)
    const float* __restrict__ bg2,
    _Float16* __restrict__ qh)       // (NGPAD, KD)
{
    const int g = blockIdx.x;
    const int k = threadIdx.x;
    if (g >= NG) { qh[g * KD + k] = (_Float16)0.0f; return; }
    __shared__ float t1[KD];
    float a = bg1[k];
    #pragma unroll 4
    for (int z = 0; z < 100; ++z)
        a = fmaf(Grep[g * 100 + z], Wg1[z * KD + k], a);
    t1[k] = gelu_exact(a);
    __syncthreads();
    float a2 = bg2[k];
    #pragma unroll
    for (int j = 0; j < KD; ++j)
        a2 = fmaf(t1[j], Wg2[j * KD + k], a2);
    qh[g * KD + k] = (_Float16)(a2 * 0.17677669529663687f);
}

// ---------------------------------------------------------------------------
// V -> f16, padded to ZPAD rows. Row 100 == 1.0 (denominator column).
// ---------------------------------------------------------------------------
__global__ __launch_bounds__(256) void vprep_kernel(
    const float* __restrict__ genZ,  // (ZD, NC)
    _Float16* __restrict__ vh)       // (ZPAD, NC)
{
    int i = blockIdx.x * 256 + threadIdx.x;
    if (i >= ZPAD * NC) return;
    int z = i / NC;
    float v = (z < ZD) ? genZ[i] : (z == ZD ? 1.0f : 0.0f);
    vh[i] = (_Float16)v;
}

// ---------------------------------------------------------------------------
// Flash attention over one cell segment; software-pipelined (2x unroll,
// double-buffered K/Gumbel register tiles; V loads issued a full tile-body
// early and pinned with a scheduling barrier). One wave owns 32 genes.
// ---------------------------------------------------------------------------
__global__ __launch_bounds__(128) void flash_kernel(
    const float* __restrict__ gumbel, // (NG, NC)
    const _Float16* __restrict__ vh,  // (ZPAD, NC)
    const _Float16* __restrict__ keyh,// (NC, KD)
    const _Float16* __restrict__ qh,  // (NGPAD, KD), pre-scaled
    float* __restrict__ pacc,         // (SEG, NGPAD, ZPAD)
    float* __restrict__ pm)           // (SEG, NGPAD)
{
    __shared__ _Float16 ldsP[4][32 * 32];

    const int wave = threadIdx.x >> 5;
    const int lane = threadIdx.x & 31;
    const int half = lane >> 4;
    const int ln   = lane & 15;
    const int nblk = NGPAD / 128;            // 79
    const int seg  = blockIdx.x / nblk;
    const int gbase = ((blockIdx.x % nblk) * 4 + wave) * 32;
    _Float16* P = ldsP[wave];

    // Q A-tiles: M=ln, K runs {half*8..+7} and {16+half*8..+7}
    v16h qa[2];
    #pragma unroll
    for (int mt = 0; mt < 2; ++mt) {
        const _Float16* qrow = qh + (size_t)(gbase + mt * 16 + ln) * KD;
        v8h lo = *(const v8h*)(qrow + half * 8);
        v8h hi = *(const v8h*)(qrow + 16 + half * 8);
        #pragma unroll
        for (int e = 0; e < 8; ++e) { qa[mt][e] = lo[e]; qa[mt][8 + e] = hi[e]; }
    }

    // Loop-invariant 32-bit per-lane gumbel offsets (elements)
    int goff[2][2][8];
    #pragma unroll
    for (int mt = 0; mt < 2; ++mt)
        #pragma unroll
        for (int nt = 0; nt < 2; ++nt)
            #pragma unroll
            for (int r = 0; r < 8; ++r) {
                int g  = gbase + mt * 16 + r + 8 * half;
                int gc = g < NG ? g : NG - 1;
                goff[mt][nt][r] = gc * NC + nt * 16 + ln;
            }

    v8f acc[2][7];                   // C layout: M=gene(r+8*half), N=z(zt*16+ln)
    float mrow[2][8];
    #pragma unroll
    for (int mt = 0; mt < 2; ++mt) {
        #pragma unroll
        for (int zt = 0; zt < 7; ++zt) acc[mt][zt] = (v8f)0.0f;
        #pragma unroll
        for (int r = 0; r < 8; ++r) mrow[mt][r] = -INFINITY;
    }

    const int T = CSEG / 32;         // 32 tiles per segment

    auto load_kb = [&](int cb, v16h kb[2]) {
        #pragma unroll
        for (int nt = 0; nt < 2; ++nt) {
            const _Float16* krow = keyh + (size_t)(cb + nt * 16 + ln) * KD + half * 8;
            v8h a = *(const v8h*)krow;
            v8h b = *(const v8h*)(krow + 16);
            #pragma unroll
            for (int e = 0; e < 8; ++e) { kb[nt][e] = a[e]; kb[nt][8 + e] = b[e]; }
        }
    };
    auto load_cg = [&](int cb, v8f cg[2][2]) {
        const float* gptr = gumbel + cb;     // uniform (SGPR) base
        #pragma unroll
        for (int mt = 0; mt < 2; ++mt)
            #pragma unroll
            for (int nt = 0; nt < 2; ++nt)
                #pragma unroll
                for (int r = 0; r < 8; ++r)
                    cg[mt][nt][r] = __builtin_nontemporal_load(gptr + goff[mt][nt][r]);
    };

    auto compute = [&](int cb, v16h kb[2], v8f cg[2][2]) {
        // --- issue ALL 14 V b128 loads first; their consumers are at the
        // bottom of this body, so the whole softmax section hides the latency.
        v16h vbt[7];
        {
            const _Float16* vbase = vh + (size_t)ln * NC + cb + half * 8;
            #pragma unroll
            for (int zt = 0; zt < 7; ++zt) {
                const _Float16* vrow = vbase + (size_t)zt * 16 * NC;
                v8h a = *(const v8h*)vrow;
                v8h b = *(const v8h*)(vrow + 16);
                #pragma unroll
                for (int e = 0; e < 8; ++e) { vbt[zt][e] = a[e]; vbt[zt][8 + e] = b[e]; }
            }
        }
        // keep the V loads up here; don't let the scheduler sink them.
        __builtin_amdgcn_sched_barrier(0);

        // scores + gumbel in one WMMA each (q pre-scaled by 1/sqrt(32))
        v8f s[2][2];
        #pragma unroll
        for (int mt = 0; mt < 2; ++mt)
            #pragma unroll
            for (int nt = 0; nt < 2; ++nt)
                s[mt][nt] = wmma_f16(qa[mt], kb[nt], cg[mt][nt]);

        #pragma unroll
        for (int mt = 0; mt < 2; ++mt) {
            // lazy online-max: rescale only if some lane beats the running max
            bool exceed = false;
            #pragma unroll
            for (int nt = 0; nt < 2; ++nt)
                #pragma unroll
                for (int r = 0; r < 8; ++r)
                    exceed |= (s[mt][nt][r] > mrow[mt][r]);
            if (__any((int)exceed)) {
                float tmax[8];
                #pragma unroll
                for (int r = 0; r < 8; ++r)
                    tmax[r] = fmaxf(s[mt][0][r], s[mt][1][r]);
                #pragma unroll
                for (int m = 1; m < 16; m <<= 1)
                    #pragma unroll
                    for (int r = 0; r < 8; ++r)
                        tmax[r] = fmaxf(tmax[r], __shfl_xor(tmax[r], m, 32));
                float corr[8];
                #pragma unroll
                for (int r = 0; r < 8; ++r) {
                    float mnew = fmaxf(mrow[mt][r], tmax[r]);
                    corr[r] = __expf(mrow[mt][r] - mnew);   // first tile: exp(-inf)=0
                    mrow[mt][r] = mnew;
                }
                #pragma unroll
                for (int zt = 0; zt < 7; ++zt)
                    #pragma unroll
                    for (int r = 0; r < 8; ++r)
                        acc[mt][zt][r] *= corr[r];
            }
            // P tile to LDS (gene-major, cells contiguous); LDS in-order per wave
            #pragma unroll
            for (int nt = 0; nt < 2; ++nt)
                #pragma unroll
                for (int r = 0; r < 8; ++r)
                    P[(mt * 16 + r + 8 * half) * 32 + nt * 16 + ln] =
                        (_Float16)__expf(s[mt][nt][r] - mrow[mt][r]);
        }

        // P A-tiles back in WMMA A layout
        v16h pa[2];
        #pragma unroll
        for (int mt = 0; mt < 2; ++mt) {
            const _Float16* prow = P + (mt * 16 + ln) * 32;
            v8h lo = *(const v8h*)(prow + half * 8);
            v8h hi = *(const v8h*)(prow + 16 + half * 8);
            #pragma unroll
            for (int e = 0; e < 8; ++e) { pa[mt][e] = lo[e]; pa[mt][8 + e] = hi[e]; }
        }

        // WMMA drain: V tiles were issued at the top of the body
        #pragma unroll
        for (int zt = 0; zt < 7; ++zt)
            #pragma unroll
            for (int mt = 0; mt < 2; ++mt)
                acc[mt][zt] = wmma_f16(pa[mt], vbt[zt], acc[mt][zt]);
    };

    // ---- software pipeline: 2x unrolled, static double buffers
    v16h kbA[2], kbB[2];
    v8f  cgA[2][2], cgB[2][2];
    const int cb0 = seg * CSEG;
    load_kb(cb0, kbA);
    load_cg(cb0, cgA);
    for (int t = 0; t < T; t += 2) {
        int cb  = seg * CSEG + t * 32;
        int tn1 = (t + 1 < T) ? t + 1 : t;        // clamped prefetch targets
        int tn2 = (t + 2 < T) ? t + 2 : T - 1;
        int cb1 = seg * CSEG + tn1 * 32;
        int cb2 = seg * CSEG + tn2 * 32;
        load_kb(cb1, kbB);
        load_cg(cb1, cgB);
        compute(cb, kbA, cgA);
        load_kb(cb2, kbA);
        load_cg(cb2, cgA);
        compute(cb1, kbB, cgB);
    }

    // --- store partials for the merge pass
    #pragma unroll
    for (int mt = 0; mt < 2; ++mt) {
        #pragma unroll
        for (int zt = 0; zt < 7; ++zt)
            #pragma unroll
            for (int r = 0; r < 8; ++r) {
                int g = gbase + mt * 16 + r + 8 * half;
                pacc[((size_t)seg * NGPAD + g) * ZPAD + zt * 16 + ln] = acc[mt][zt][r];
            }
        if (ln == 0) {
            #pragma unroll
            for (int r = 0; r < 8; ++r)
                pm[(size_t)seg * NGPAD + gbase + mt * 16 + r + 8 * half] = mrow[mt][r];
        }
    }
}

// ---------------------------------------------------------------------------
// Merge SEG partials: online-softmax combine, normalize, transpose-store.
// ---------------------------------------------------------------------------
__global__ __launch_bounds__(128) void merge_kernel(
    const float* __restrict__ pacc,   // (SEG, NGPAD, ZPAD)
    const float* __restrict__ pm,     // (SEG, NGPAD)
    float* __restrict__ out)          // (ZD, NG)
{
    int g = blockIdx.x * 128 + threadIdx.x;
    if (g >= NG) return;
    float m = -INFINITY;
    #pragma unroll
    for (int s = 0; s < SEG; ++s) m = fmaxf(m, pm[(size_t)s * NGPAD + g]);
    float w[SEG];
    float denom = 0.0f;
    #pragma unroll
    for (int s = 0; s < SEG; ++s) {
        w[s] = __expf(pm[(size_t)s * NGPAD + g] - m);
        denom = fmaf(pacc[((size_t)s * NGPAD + g) * ZPAD + ZD], w[s], denom);
    }
    float inv = 1.0f / denom;
    for (int z = 0; z < ZD; ++z) {
        float num = 0.0f;
        #pragma unroll
        for (int s = 0; s < SEG; ++s)
            num = fmaf(pacc[((size_t)s * NGPAD + g) * ZPAD + z], w[s], num);
        out[(size_t)z * NG + g] = num * inv;
    }
}

// ---------------------------------------------------------------------------
extern "C" void kernel_launch(void* const* d_in, const int* in_sizes, int n_in,
                              void* d_out, int out_size, void* d_ws, size_t ws_size,
                              hipStream_t stream) {
    const float* rawZ   = (const float*)d_in[0];
    const float* genZ   = (const float*)d_in[1];
    const float* Grep   = (const float*)d_in[2];
    const float* gumbel = (const float*)d_in[3];
    const float* Wz1    = (const float*)d_in[4];
    const float* bz1    = (const float*)d_in[5];
    const float* Wz2    = (const float*)d_in[6];
    const float* bz2    = (const float*)d_in[7];
    const float* Wg1    = (const float*)d_in[8];
    const float* bg1    = (const float*)d_in[9];
    const float* Wg2    = (const float*)d_in[10];
    const float* bg2    = (const float*)d_in[11];
    float* out = (float*)d_out;

    // workspace layout (~40 MB total)
    _Float16* keyh = (_Float16*)d_ws;                    // NC*KD        f16
    _Float16* qh   = keyh + (size_t)NC * KD;             // NGPAD*KD     f16
    _Float16* vh   = qh + (size_t)NGPAD * KD;            // ZPAD*NC      f16
    float*    pacc = (float*)(vh + (size_t)ZPAD * NC);   // SEG*NGPAD*ZPAD f32
    float*    pm   = pacc + (size_t)SEG * NGPAD * ZPAD;  // SEG*NGPAD    f32

    keyz_kernel<<<NC, HD, 0, stream>>>(rawZ, Wz1, bz1, Wz2, bz2, keyh);
    queryg_kernel<<<NGPAD, KD, 0, stream>>>(Grep, Wg1, bg1, Wg2, bg2, qh);
    vprep_kernel<<<(ZPAD * NC + 255) / 256, 256, 0, stream>>>(genZ, vh);
    flash_kernel<<<(NGPAD / 128) * SEG, 128, 0, stream>>>(gumbel, vh, keyh, qh, pacc, pm);
    merge_kernel<<<(NG + 127) / 128, 128, 0, stream>>>(pacc, pm, out);
}
